// HeroGRAPH_91216515432634
// MI455X (gfx1250) — matrix operand order repaired and checked by reading, not compiled, not run on hardware
//
#include <hip/hip_runtime.h>

// ---------------------------------------------------------------------------
// Bipartite 2-layer GAT (edge-softmax factored through per-node scalars) +
// WMMA f16->f32 node GEMMs for V2 / upd1 / upd2 / fc on gfx1250 (wave32).
// ---------------------------------------------------------------------------

typedef _Float16 half_t;
typedef _Float16 hv8  __attribute__((ext_vector_type(8)));
typedef _Float16 hv16 __attribute__((ext_vector_type(16)));
typedef float    fv8  __attribute__((ext_vector_type(8)));

#define D64 64

__device__ __forceinline__ float sigf(float x) { return 1.0f / (1.0f + __expf(-x)); }

// -------- f32 -> f16 convert (weights) --------
__global__ void k_cvt_f16(const float* __restrict__ s, half_t* __restrict__ d, int n) {
    int i = blockIdx.x * blockDim.x + threadIdx.x;
    if (i < n) d[i] = (half_t)s[i];
}

// -------- per-node prep: attention scalars + f16 feature copy --------
__global__ void k_node_prep(const float* __restrict__ q,
                            const float* __restrict__ wA,   // W_att[0:64]
                            const float* __restrict__ wB,   // W_att[64:128]
                            float* __restrict__ sK, float* __restrict__ sQ,
                            half_t* __restrict__ q16, int n) {
    int i = blockIdx.x * blockDim.x + threadIdx.x;
    if (i >= n) return;
    const float* row = q + (size_t)i * D64;
    float a = 0.f, b = 0.f;
#pragma unroll
    for (int k = 0; k < D64; k++) {
        float v = row[k];
        a = __builtin_fmaf(v, wA[k], a);
        b = __builtin_fmaf(v, wB[k], b);
        q16[(size_t)i * D64 + k] = (half_t)v;
    }
    sK[i] = a;
    sQ[i] = b;
}

// -------- edge pass 1: segment max of sigmoid logits (both directions) -----
// sigmoid > 0, so uint-bit atomicMax on positive floats is exact; init = 0.
__global__ void k_edge_max(const int* __restrict__ src, const int* __restrict__ dst,
                           const float* __restrict__ sKu, const float* __restrict__ sQu,
                           const float* __restrict__ sKi, const float* __restrict__ sQi,
                           unsigned* __restrict__ m_i, unsigned* __restrict__ m_u, int ne) {
    int e = blockIdx.x * blockDim.x + threadIdx.x;
    if (e >= ne) return;
    int s = src[e], d = dst[e];
    atomicMax(&m_i[d], __float_as_uint(sigf(sKu[s] + sQi[d])));
    atomicMax(&m_u[s], __float_as_uint(sigf(sKi[d] + sQu[s])));
}

// -------- edge pass 2: exp-weighted scatter of features (+V), denominator --
// 64 threads per edge, one per feature; OUT==0 skips the OV path (layer 1).
__global__ void k_edge_acc(const int* __restrict__ src, const int* __restrict__ dst,
                           const float* __restrict__ sKu, const float* __restrict__ sQu,
                           const float* __restrict__ sKi, const float* __restrict__ sQi,
                           const float* __restrict__ m_i, const float* __restrict__ m_u,
                           const float* __restrict__ qu, const float* __restrict__ qi,
                           const float* __restrict__ Vu, const float* __restrict__ Vi,
                           float* __restrict__ den_i, float* __restrict__ den_u,
                           float* __restrict__ OKi, float* __restrict__ OKu,
                           float* __restrict__ OVi, float* __restrict__ OVu,
                           int ne, int OUT) {
    long long t = (long long)blockIdx.x * blockDim.x + threadIdx.x;
    int e = (int)(t >> 6), f = (int)(t & 63);
    if (e >= ne) return;
    int s = src[e], d = dst[e];

    float exui = __expf(sigf(sKu[s] + sQi[d]) - m_i[d]);   // user -> item
    float exiu = __expf(sigf(sKi[d] + sQu[s]) - m_u[s]);   // item -> user
    if (f == 0) {
        atomicAdd(&den_i[d], exui);
        atomicAdd(&den_u[s], exiu);
    }
    atomicAdd(&OKi[(size_t)d * D64 + f], exui * qu[(size_t)s * D64 + f]);
    atomicAdd(&OKu[(size_t)s * D64 + f], exiu * qi[(size_t)d * D64 + f]);
    if (f < OUT) {
        atomicAdd(&OVi[(size_t)d * OUT + f], exui * Vu[(size_t)s * OUT + f]);
        atomicAdd(&OVu[(size_t)s * OUT + f], exiu * Vi[(size_t)d * OUT + f]);
    }
}

// -------- normalize + residual: A16 = f16(q + OK/denom) --------
__global__ void k_finish(const float* __restrict__ q, const float* __restrict__ OK,
                         const float* __restrict__ den, half_t* __restrict__ A16, int n) {
    long long t = (long long)blockIdx.x * blockDim.x + threadIdx.x;
    int i = (int)(t >> 6);
    if (i >= n) return;
    float inv = 1.0f / fmaxf(den[i], 1e-12f);
    A16[t] = (half_t)(q[t] + OK[t] * inv);
}

// -------- pack [q2 | OV2/denom] as f16 [N,32] for the fc GEMM --------
__global__ void k_pack_concat(const float* __restrict__ q2, const float* __restrict__ OVraw,
                              const float* __restrict__ den, half_t* __restrict__ cat, int n) {
    long long t = (long long)blockIdx.x * blockDim.x + threadIdx.x;
    int i = (int)(t >> 4), f = (int)(t & 15);
    if (i >= n) return;
    float inv = 1.0f / fmaxf(den[i], 1e-12f);
    cat[(size_t)i * 32 + f]      = (half_t)q2[(size_t)i * 16 + f];
    cat[(size_t)i * 32 + 16 + f] = (half_t)(OVraw[(size_t)i * 16 + f] * inv);
}

// -------- out[:, 0:64] = embeddings --------
__global__ void k_copy_embed(const float* __restrict__ Eemb, float* __restrict__ out, int n) {
    long long t = (long long)blockIdx.x * blockDim.x + threadIdx.x;
    int i = (int)(t >> 6), f = (int)(t & 63);
    if (i >= n) return;
    out[(size_t)i * 128 + f] = Eemb[(size_t)i * 64 + f];
}

// ---------------------------------------------------------------------------
// C[N,No] = act( A[N,Ki] @ W[No,Ki]^T + bias ), one wave per 16x16 tile,
// v_wmma_f32_16x16x32_f16, K stepped by 32.  N must be a multiple of 16
// (N = 50000 = 3125*16 here).  Tile predicate is wave-uniform so EXEC is
// all-ones around the WMMA as the ISA requires.
// ---------------------------------------------------------------------------
__global__ void k_wmma_gemm_nt(const half_t* __restrict__ A, int lda,
                               const half_t* __restrict__ W, int ldw,
                               const float* __restrict__ bias,
                               float* __restrict__ C, int ldc, int coff,
                               int N, int Ki, int No, int relu) {
    int wavesPerBlock = blockDim.x >> 5;
    int wave = blockIdx.x * wavesPerBlock + (threadIdx.x >> 5);
    int lane = threadIdx.x & 31;
    int colTiles = No >> 4;
    int rowTiles = N >> 4;
    if (wave >= rowTiles * colTiles) return;   // wave-uniform exit
    int rowT = wave / colTiles, colT = wave % colTiles;
    int h  = lane >> 4;        // half-wave select
    int rm = lane & 15;        // row within A tile / row of W (output column)

    const half_t* arow = A + (size_t)(rowT * 16 + rm) * lda;
    const half_t* wrow = W + (size_t)(colT * 16 + rm) * ldw;

    fv8 acc = {};
    for (int k0 = 0; k0 < Ki; k0 += 32) {
        // A 16x32 f16 layout: lane half h holds K = {h*8..h*8+7, 16+h*8..16+h*8+7}
        hv8 alo = *(const hv8*)(arow + k0 + h * 8);
        hv8 ahi = *(const hv8*)(arow + k0 + 16 + h * 8);
        // B 32x16 f16 layout: lane half h holds K = {h*8..h*8+7, 16+h*8..16+h*8+7} of column rm
        hv8 blo = *(const hv8*)(wrow + k0 + h * 8);
        hv8 bhi = *(const hv8*)(wrow + k0 + 16 + h * 8);
        hv16 a = __builtin_shufflevector(alo, ahi, 0,1,2,3,4,5,6,7,8,9,10,11,12,13,14,15);
        hv16 b = __builtin_shufflevector(blo, bhi, 0,1,2,3,4,5,6,7,8,9,10,11,12,13,14,15);
        acc = __builtin_amdgcn_wmma_f32_16x16x32_f16(false, a, false, b,
                                                     (short)0, acc, false, false);
    }
    // D layout: VGPR j -> row rowT*16 + h*8 + j, column colT*16 + rm
    int ncol = colT * 16 + rm;
    float badd = bias ? bias[ncol] : 0.0f;
#pragma unroll
    for (int j = 0; j < 8; j++) {
        int m = rowT * 16 + h * 8 + j;
        float v = acc[j] + badd;
        if (relu) v = fmaxf(v, 0.0f);
        C[(size_t)m * ldc + coff + ncol] = v;
    }
}

// ---------------------------------------------------------------------------
extern "C" void kernel_launch(void* const* d_in, const int* in_sizes, int n_in,
                              void* d_out, int out_size, void* d_ws, size_t ws_size,
                              hipStream_t stream) {
    const float* E_u    = (const float*)d_in[0];
    const float* E_i    = (const float*)d_in[1];
    const float* Q_u    = (const float*)d_in[2];
    const float* Q_i    = (const float*)d_in[3];
    const float* W_att1 = (const float*)d_in[4];
    const float* W_upd1 = (const float*)d_in[5];
    // d_in[6] = W_V1, d_in[7] = b_V1 : layer-1 OV is discarded by the reference
    const float* W_att2 = (const float*)d_in[8];
    const float* W_upd2 = (const float*)d_in[9];
    const float* W_V2   = (const float*)d_in[10];
    const float* b_V2   = (const float*)d_in[11];
    const float* W_fc   = (const float*)d_in[12];
    const float* b_fc   = (const float*)d_in[13];
    const int*   src    = (const int*)d_in[14];
    const int*   dst    = (const int*)d_in[15];

    const int NU = in_sizes[0] / 64;
    const int NI = in_sizes[1] / 64;
    const int NE = in_sizes[14];

    // ---- workspace carve-out ----
    char* ws = (char*)d_ws;
    size_t off = 0;
    auto alloc = [&](size_t bytes) -> void* {
        void* p = ws + off;
        off = (off + bytes + 255) & ~(size_t)255;
        return p;
    };
    half_t* w_upd1_16 = (half_t*)alloc(64 * 64 * 2);
    half_t* w_upd2_16 = (half_t*)alloc(16 * 64 * 2);
    half_t* w_v2_16   = (half_t*)alloc(16 * 64 * 2);
    half_t* w_fc_16   = (half_t*)alloc(64 * 32 * 2);
    float* sKu = (float*)alloc((size_t)NU * 4);
    float* sQu = (float*)alloc((size_t)NU * 4);
    float* sKi = (float*)alloc((size_t)NI * 4);
    float* sQi = (float*)alloc((size_t)NI * 4);
    float* m_u = (float*)alloc((size_t)NU * 4);
    float* m_i = (float*)alloc((size_t)NI * 4);
    float* den_u = (float*)alloc((size_t)NU * 4);
    float* den_i = (float*)alloc((size_t)NI * 4);
    float* OKu = (float*)alloc((size_t)NU * 64 * 4);
    float* OKi = (float*)alloc((size_t)NI * 64 * 4);
    float* OVu = (float*)alloc((size_t)NU * 16 * 4);
    float* OVi = (float*)alloc((size_t)NI * 16 * 4);
    half_t* qu16 = (half_t*)alloc((size_t)NU * 64 * 2);
    half_t* qi16 = (half_t*)alloc((size_t)NI * 64 * 2);
    half_t* A16u = (half_t*)alloc((size_t)NU * 64 * 2);
    half_t* A16i = (half_t*)alloc((size_t)NI * 64 * 2);
    float* q1u = (float*)alloc((size_t)NU * 64 * 4);
    float* q1i = (float*)alloc((size_t)NI * 64 * 4);
    float* V2u = (float*)alloc((size_t)NU * 16 * 4);
    float* V2i = (float*)alloc((size_t)NI * 16 * 4);
    float* q2u = (float*)alloc((size_t)NU * 16 * 4);
    float* q2i = (float*)alloc((size_t)NI * 16 * 4);
    half_t* catu = (half_t*)alloc((size_t)NU * 32 * 2);
    half_t* cati = (half_t*)alloc((size_t)NI * 32 * 2);

    const int B = 256;
    auto cdiv = [](long long a, long long b) -> int { return (int)((a + b - 1) / b); };
    auto gemm = [&](const half_t* A, int lda, const half_t* W, int ldw, const float* bias,
                    float* C, int ldc, int coff, int N, int Ki, int No, int relu) {
        long long waves = (long long)(N / 16) * (No / 16);
        k_wmma_gemm_nt<<<cdiv(waves * 32, B), B, 0, stream>>>(A, lda, W, ldw, bias, C, ldc,
                                                              coff, N, Ki, No, relu);
    };

    // ---- weight conversions (deterministic every call) ----
    k_cvt_f16<<<cdiv(64 * 64, B), B, 0, stream>>>(W_upd1, w_upd1_16, 64 * 64);
    k_cvt_f16<<<cdiv(16 * 64, B), B, 0, stream>>>(W_upd2, w_upd2_16, 16 * 64);
    k_cvt_f16<<<cdiv(16 * 64, B), B, 0, stream>>>(W_V2, w_v2_16, 16 * 64);
    k_cvt_f16<<<cdiv(64 * 32, B), B, 0, stream>>>(W_fc, w_fc_16, 64 * 32);

    // ================= layer 1 (d=64, OV discarded) =================
    k_node_prep<<<cdiv(NU, B), B, 0, stream>>>(Q_u, W_att1, W_att1 + 64, sKu, sQu, qu16, NU);
    k_node_prep<<<cdiv(NI, B), B, 0, stream>>>(Q_i, W_att1, W_att1 + 64, sKi, sQi, qi16, NI);
    hipMemsetAsync(m_u, 0, (size_t)NU * 4, stream);
    hipMemsetAsync(m_i, 0, (size_t)NI * 4, stream);
    hipMemsetAsync(den_u, 0, (size_t)NU * 4, stream);
    hipMemsetAsync(den_i, 0, (size_t)NI * 4, stream);
    hipMemsetAsync(OKu, 0, (size_t)NU * 64 * 4, stream);
    hipMemsetAsync(OKi, 0, (size_t)NI * 64 * 4, stream);
    k_edge_max<<<cdiv(NE, B), B, 0, stream>>>(src, dst, sKu, sQu, sKi, sQi,
                                              (unsigned*)m_i, (unsigned*)m_u, NE);
    k_edge_acc<<<cdiv((long long)NE * 64, B), B, 0, stream>>>(
        src, dst, sKu, sQu, sKi, sQi, m_i, m_u, Q_u, Q_i, nullptr, nullptr,
        den_i, den_u, OKi, OKu, nullptr, nullptr, NE, 0);
    k_finish<<<cdiv((long long)NU * 64, B), B, 0, stream>>>(Q_u, OKu, den_u, A16u, NU);
    k_finish<<<cdiv((long long)NI * 64, B), B, 0, stream>>>(Q_i, OKi, den_i, A16i, NI);
    gemm(A16u, 64, w_upd1_16, 64, nullptr, q1u, 64, 0, NU, 64, 64, 0);
    gemm(A16i, 64, w_upd1_16, 64, nullptr, q1i, 64, 0, NI, 64, 64, 0);

    // ================= layer 2 (d=64 -> 16, with OV) =================
    k_node_prep<<<cdiv(NU, B), B, 0, stream>>>(q1u, W_att2, W_att2 + 64, sKu, sQu, qu16, NU);
    k_node_prep<<<cdiv(NI, B), B, 0, stream>>>(q1i, W_att2, W_att2 + 64, sKi, sQi, qi16, NI);
    gemm(qu16, 64, w_v2_16, 64, b_V2, V2u, 16, 0, NU, 64, 16, 0);
    gemm(qi16, 64, w_v2_16, 64, b_V2, V2i, 16, 0, NI, 64, 16, 0);
    hipMemsetAsync(m_u, 0, (size_t)NU * 4, stream);
    hipMemsetAsync(m_i, 0, (size_t)NI * 4, stream);
    hipMemsetAsync(den_u, 0, (size_t)NU * 4, stream);
    hipMemsetAsync(den_i, 0, (size_t)NI * 4, stream);
    hipMemsetAsync(OKu, 0, (size_t)NU * 64 * 4, stream);
    hipMemsetAsync(OKi, 0, (size_t)NI * 64 * 4, stream);
    hipMemsetAsync(OVu, 0, (size_t)NU * 16 * 4, stream);
    hipMemsetAsync(OVi, 0, (size_t)NI * 16 * 4, stream);
    k_edge_max<<<cdiv(NE, B), B, 0, stream>>>(src, dst, sKu, sQu, sKi, sQi,
                                              (unsigned*)m_i, (unsigned*)m_u, NE);
    k_edge_acc<<<cdiv((long long)NE * 64, B), B, 0, stream>>>(
        src, dst, sKu, sQu, sKi, sQi, m_i, m_u, q1u, q1i, V2u, V2i,
        den_i, den_u, OKi, OKu, OVi, OVu, NE, 16);
    k_finish<<<cdiv((long long)NU * 64, B), B, 0, stream>>>(q1u, OKu, den_u, A16u, NU);
    k_finish<<<cdiv((long long)NI * 64, B), B, 0, stream>>>(q1i, OKi, den_i, A16i, NI);
    gemm(A16u, 64, w_upd2_16, 64, nullptr, q2u, 16, 0, NU, 64, 16, 0);
    gemm(A16i, 64, w_upd2_16, 64, nullptr, q2i, 16, 0, NI, 64, 16, 0);

    // ================= head: concat -> fc(relu) -> output =================
    float* out = (float*)d_out;
    k_pack_concat<<<cdiv((long long)NU * 16, B), B, 0, stream>>>(q2u, OVu, den_u, catu, NU);
    k_pack_concat<<<cdiv((long long)NI * 16, B), B, 0, stream>>>(q2i, OVi, den_i, cati, NI);
    gemm(catu, 32, w_fc_16, 32, b_fc, out, 128, 64, NU, 32, 64, 1);
    gemm(cati, 32, w_fc_16, 32, b_fc, out + (size_t)NU * 128, 128, 64, NI, 32, 64, 1);
    k_copy_embed<<<cdiv((long long)NU * 64, B), B, 0, stream>>>(E_u, out, NU);
    k_copy_embed<<<cdiv((long long)NI * 64, B), B, 0, stream>>>(E_i, out + (size_t)NU * 128, NI);
}
